// MultiHeadAttention_84834194030917
// MI455X (gfx1250) — matrix operand order repaired and checked by reading, not compiled, not run on hardware
//
#include <hip/hip_runtime.h>
#include <hip/hip_bf16.h>
#include <stdint.h>

// ---------------------------------------------------------------------------
// MHA on gfx1250: bf16 WMMA GEMMs + flash-style SDPA with async-LDS staging.
//   B=2, S=2048, D=1024, H=16, dk=64
// ---------------------------------------------------------------------------

typedef __attribute__((ext_vector_type(16))) __bf16 bf16x16;
typedef __attribute__((ext_vector_type(8)))  __bf16 bf16x8;
typedef __attribute__((ext_vector_type(8)))  float  f32x8;
typedef int v4i __attribute__((vector_size(16)));   // matches builtin pointee

constexpr int BATCH  = 2;
constexpr int SEQ    = 2048;
constexpr int DMODEL = 1024;
constexpr int HEADS  = 16;
constexpr int DK     = 64;
constexpr int MROWS  = BATCH * SEQ;          // 4096

// ---------------------------------------------------------------------------
// Async global->LDS copy (16B per lane) with safe fallback.
// Builtin signature (per hipcc diagnostic): param0 = v4i addrspace(1)*,
// param1 = v4i addrspace(3)*, then imm offset, imm cpol.
// Flat addresses of LDS carry the LDS byte offset in bits[31:0] (ISA 10.2),
// so the uintptr_t round-trip casts are numerically correct.
// ---------------------------------------------------------------------------
#if defined(__gfx1250__) && __has_builtin(__builtin_amdgcn_global_load_async_to_lds_b128)
#define HAS_ASYNC_LDS 1
#else
#define HAS_ASYNC_LDS 0
#endif

__device__ inline void copy16_g2lds(const __bf16* gsrc, __bf16* ldst) {
#if HAS_ASYNC_LDS
  __builtin_amdgcn_global_load_async_to_lds_b128(
      (__attribute__((address_space(1))) v4i*)(uintptr_t)gsrc,
      (__attribute__((address_space(3))) v4i*)(uint32_t)(uintptr_t)ldst,
      0, 0);
#else
  *(uint4*)ldst = *(const uint4*)gsrc;
#endif
}

__device__ inline void async_copy_wait() {
#if HAS_ASYNC_LDS
# if __has_builtin(__builtin_amdgcn_s_wait_asynccnt)
  __builtin_amdgcn_s_wait_asynccnt(0);
# else
  asm volatile("s_wait_asynccnt 0x0" ::: "memory");
# endif
#endif
}

// ---------------------------------------------------------------------------
// Fragment loaders (CDNA5 wave32 WMMA 16x16x32 bf16 layouts, ISA 7.12.2)
// A (16x32, MxK): lane<16 -> row=lane,    K = {0..7, 16..23}
//                 lane>=16 -> row=lane-16, K = {8..15, 24..31}
// B (32x16, KxN): lane<16 -> col=lane,    K = {0..7, 16..23}
//                 lane>=16 -> col=lane-16, K = {8..15, 24..31}
// C/D (16x16):    vgpr r, lane<16 -> (M=r,  N=lane)
//                 vgpr r, lane>=16 -> (M=r+8, N=lane-16)
// ---------------------------------------------------------------------------

__device__ inline bf16x16 load_a_frag(const __bf16* __restrict__ A, int lda,
                                      int row0, int k0, int lane) {
  const int row = row0 + (lane & 15);
  const int kb  = k0 + ((lane >> 4) << 3);      // 0 or 8
  const __bf16* p = A + (size_t)row * lda + kb;
  bf16x8 lo = *(const bf16x8*)(p);
  bf16x8 hi = *(const bf16x8*)(p + 16);
  bf16x16 f;
#pragma unroll
  for (int i = 0; i < 8; ++i) { f[i] = lo[i]; f[8 + i] = hi[i]; }
  return f;
}

// B-fragment where B[k][n] = Mem[n][k] (column n of B is row n of Mem).
// Works for x@W^T, K-tiles of Q.K^T, transposed-V tiles, and LDS tiles.
__device__ inline bf16x16 load_bt_frag(const __bf16* __restrict__ Bm, int ldb,
                                       int col0, int k0, int lane) {
  const int col = col0 + (lane & 15);
  const int kb  = k0 + ((lane >> 4) << 3);      // 0 or 8
  const __bf16* p = Bm + (size_t)col * ldb + kb;
  bf16x8 lo = *(const bf16x8*)(p);
  bf16x8 hi = *(const bf16x8*)(p + 16);
  bf16x16 f;
#pragma unroll
  for (int i = 0; i < 8; ++i) { f[i] = lo[i]; f[8 + i] = hi[i]; }
  return f;
}

// ---------------------------------------------------------------------------
// fp32 -> bf16 conversion
// ---------------------------------------------------------------------------
__global__ void cvt_f32_to_bf16(const float* __restrict__ x,
                                __bf16* __restrict__ y, int n) {
  int i = blockIdx.x * blockDim.x + threadIdx.x;
  if (i < n) y[i] = (__bf16)x[i];
}

// ---------------------------------------------------------------------------
// GEMM: Out = A[M,K] @ W[N,K]^T  (bf16 in, fp32 accumulate via WMMA)
// mode 0: write bf16 split-heads   [B,H,S,dk]
// mode 1: write bf16 split-heads-T [B,H,dk,S]
// mode 2: write fp32 row-major     [M,N]
// One wave -> 32x64 tile (8 accumulators; 8 WMMA / 12 b128-loads per K-step).
// ---------------------------------------------------------------------------
__global__ __launch_bounds__(256)
void gemm_bf16_wmma(const __bf16* __restrict__ A, const __bf16* __restrict__ W,
                    void* __restrict__ Out, int M, int N, int K, int mode) {
  const int lane = threadIdx.x & 31;
  const int wave = threadIdx.x >> 5;
  const int flat = blockIdx.x * 8 + wave;
  const int ntiles = N >> 6;                       // N/64
  const int mt = flat / ntiles;
  const int nt = flat - mt * ntiles;
  if (mt >= (M >> 5)) return;                      // wave-uniform

  f32x8 acc[8] = {};

  for (int k0 = 0; k0 < K; k0 += 32) {
    __builtin_prefetch(A + (size_t)(mt * 32 + (lane & 15)) * K + k0 + 128, 0, 1);
    bf16x16 a0 = load_a_frag(A, K, mt * 32,      k0, lane);
    bf16x16 a1 = load_a_frag(A, K, mt * 32 + 16, k0, lane);
#pragma unroll
    for (int j = 0; j < 4; ++j) {
      bf16x16 b = load_bt_frag(W, K, nt * 64 + j * 16, k0, lane);
      acc[j]     = __builtin_amdgcn_wmma_f32_16x16x32_bf16(false, a0, false, b, (short)0, acc[j],     false, false);
      acc[4 + j] = __builtin_amdgcn_wmma_f32_16x16x32_bf16(false, a1, false, b, (short)0, acc[4 + j], false, false);
    }
  }

  const int ln = lane & 15;
#pragma unroll
  for (int sub = 0; sub < 2; ++sub) {
    const int mbase = mt * 32 + sub * 16 + ((lane >> 4) << 3);
#pragma unroll
    for (int j = 0; j < 4; ++j) {
      const int gcol = nt * 64 + j * 16 + ln;
#pragma unroll
      for (int r = 0; r < 8; ++r) {
        const int gr = mbase + r;
        const float v = acc[sub * 4 + j][r];
        if (mode == 2) {
          ((float*)Out)[(size_t)gr * N + gcol] = v;
        } else {
          const int b = gr / SEQ, s = gr - b * SEQ;
          const int h = gcol >> 6, d = gcol & 63;
          size_t addr;
          if (mode == 0) addr = (((size_t)(b * HEADS + h) * SEQ) + s) * DK + d;
          else           addr = (((size_t)(b * HEADS + h) * DK) + d) * SEQ + s;
          ((__bf16*)Out)[addr] = (__bf16)v;
        }
      }
    }
  }
}

// ---------------------------------------------------------------------------
// Flash-style SDPA. Block = 8 waves covering 8 q-tiles of ONE (b,h):
// K/V tiles for each 32-key block are staged once into LDS (async load to
// LDS), shared by all 8 waves. Per 32-key block per wave:
// 4 WMMA (QK^T) + online softmax + 4 WMMA (PV).
// Qh,Kh: [B,H,S,dk] bf16 ; Vt: [B,H,dk,S] bf16 ; Out: [B,S,D] bf16.
// ---------------------------------------------------------------------------
__global__ __launch_bounds__(256)
void attn_wmma(const __bf16* __restrict__ Qh, const __bf16* __restrict__ Kh,
               const __bf16* __restrict__ Vt, __bf16* __restrict__ Outp) {
  __shared__ __align__(16) __bf16 Ktile[32 * 64];   // [k 0..31][d 0..63]
  __shared__ __align__(16) __bf16 Vtile[64 * 32];   // [d 0..63][k 0..31]
  __shared__ __align__(16) __bf16 Ptile[8][16 * 32];

  const int lane = threadIdx.x & 31;
  const int wave = threadIdx.x >> 5;
  const int bh = blockIdx.x >> 4;                  // b*H + h, 0..31
  const int qt = (blockIdx.x & 15) * 8 + wave;     // query tile 0..127

  const __bf16* Qb = Qh + (size_t)bh * SEQ * DK;
  const __bf16* Kb = Kh + (size_t)bh * SEQ * DK;
  const __bf16* Vb = Vt + (size_t)bh * DK * SEQ;

  // Q fragments for the two 32-wide halves of dk=64 (held for whole loop)
  const bf16x16 qf0 = load_a_frag(Qb, DK, qt * 16, 0, lane);
  const bf16x16 qf1 = load_a_frag(Qb, DK, qt * 16, 32, lane);

  f32x8 o0 = {}, o1 = {}, o2 = {}, o3 = {};
  f32x8 mrow, lrow = {};
#pragma unroll
  for (int r = 0; r < 8; ++r) mrow[r] = -3.0e38f;

  const int pm = (lane >> 4) << 3;                 // P write row base (0|8)
  const int pn = lane & 15;                        // P write col
  const int rm = lane & 15;                        // P read row
  const int rk = (lane >> 4) << 3;                 // P read K base (0|8)

  const int t = threadIdx.x;
  // staging assignments: 256 threads x 16B cover each 4KB tile exactly
  const int krow = t >> 3, kcol = (t & 7) << 3;    // K: 32 rows x 64 cols
  const int vrow = t >> 2, vcol = (t & 3) << 3;    // V: 64 rows x 32 cols

  for (int kb = 0; kb < SEQ; kb += 32) {
    // ---- cooperative K/V staging into LDS ---------------------------------
    copy16_g2lds(Kb + (size_t)(kb + krow) * DK + kcol, &Ktile[krow * 64 + kcol]);
    copy16_g2lds(Vb + (size_t)vrow * SEQ + kb + vcol,  &Vtile[vrow * 32 + vcol]);
    async_copy_wait();
    __syncthreads();

    // ---- scores: two 16x16 tiles over this 32-key block -------------------
    f32x8 s0 = {}, s1 = {};
    {
      bf16x16 b00 = load_bt_frag(Ktile, DK, 0,   0, lane);
      bf16x16 b01 = load_bt_frag(Ktile, DK, 0,  32, lane);
      s0 = __builtin_amdgcn_wmma_f32_16x16x32_bf16(false, qf0, false, b00, (short)0, s0, false, false);
      s0 = __builtin_amdgcn_wmma_f32_16x16x32_bf16(false, qf1, false, b01, (short)0, s0, false, false);
      bf16x16 b10 = load_bt_frag(Ktile, DK, 16,  0, lane);
      bf16x16 b11 = load_bt_frag(Ktile, DK, 16, 32, lane);
      s1 = __builtin_amdgcn_wmma_f32_16x16x32_bf16(false, qf0, false, b10, (short)0, s1, false, false);
      s1 = __builtin_amdgcn_wmma_f32_16x16x32_bf16(false, qf1, false, b11, (short)0, s1, false, false);
    }
#pragma unroll
    for (int r = 0; r < 8; ++r) { s0[r] *= 0.125f; s1[r] *= 0.125f; }

    // ---- online softmax ---------------------------------------------------
    f32x8 mnew, alpha;
#pragma unroll
    for (int r = 0; r < 8; ++r) {
      float v = fmaxf(s0[r], s1[r]);
      v = fmaxf(v, __shfl_xor(v, 1));
      v = fmaxf(v, __shfl_xor(v, 2));
      v = fmaxf(v, __shfl_xor(v, 4));
      v = fmaxf(v, __shfl_xor(v, 8));
      mnew[r]  = fmaxf(mrow[r], v);
      alpha[r] = __expf(mrow[r] - mnew[r]);
      mrow[r]  = mnew[r];
    }
#pragma unroll
    for (int r = 0; r < 8; ++r) {
      s0[r] = __expf(s0[r] - mnew[r]);
      s1[r] = __expf(s1[r] - mnew[r]);
      float v = s0[r] + s1[r];
      v += __shfl_xor(v, 1);
      v += __shfl_xor(v, 2);
      v += __shfl_xor(v, 4);
      v += __shfl_xor(v, 8);
      lrow[r] = lrow[r] * alpha[r] + v;
      o0[r] *= alpha[r]; o1[r] *= alpha[r]; o2[r] *= alpha[r]; o3[r] *= alpha[r];
    }

    // ---- relayout P (D-layout -> A-layout) through per-wave LDS -----------
#pragma unroll
    for (int r = 0; r < 8; ++r) {
      Ptile[wave][(pm + r) * 32 + pn]      = (__bf16)s0[r];
      Ptile[wave][(pm + r) * 32 + pn + 16] = (__bf16)s1[r];
    }
    asm volatile("s_wait_dscnt 0x0" ::: "memory");
    bf16x16 pf;
    {
      const __bf16* lp = &Ptile[wave][rm * 32 + rk];
      bf16x8 lo = *(const bf16x8*)(lp);
      bf16x8 hi = *(const bf16x8*)(lp + 16);
#pragma unroll
      for (int i = 0; i < 8; ++i) { pf[i] = lo[i]; pf[8 + i] = hi[i]; }
    }
    asm volatile("s_wait_dscnt 0x0" ::: "memory");

    // ---- O += P @ V  (V fragments straight from LDS tile) -----------------
    bf16x16 v0 = load_bt_frag(Vtile, 32,  0, 0, lane);
    bf16x16 v1 = load_bt_frag(Vtile, 32, 16, 0, lane);
    bf16x16 v2 = load_bt_frag(Vtile, 32, 32, 0, lane);
    bf16x16 v3 = load_bt_frag(Vtile, 32, 48, 0, lane);
    o0 = __builtin_amdgcn_wmma_f32_16x16x32_bf16(false, pf, false, v0, (short)0, o0, false, false);
    o1 = __builtin_amdgcn_wmma_f32_16x16x32_bf16(false, pf, false, v1, (short)0, o1, false, false);
    o2 = __builtin_amdgcn_wmma_f32_16x16x32_bf16(false, pf, false, v2, (short)0, o2, false, false);
    o3 = __builtin_amdgcn_wmma_f32_16x16x32_bf16(false, pf, false, v3, (short)0, o3, false, false);

    __syncthreads();   // protect K/V tiles before next staging round
  }

  // ---- normalize + store concat-heads bf16 [B,S,D] ------------------------
  const int b = bh / HEADS, h = bh - b * HEADS;
  const int mb = (lane >> 4) << 3;
  const int ln = lane & 15;
  f32x8 os[4] = {o0, o1, o2, o3};
#pragma unroll
  for (int j = 0; j < 4; ++j) {
#pragma unroll
    for (int r = 0; r < 8; ++r) {
      const int s = qt * 16 + mb + r;
      const int d = h * DK + j * 16 + ln;
      Outp[((size_t)(b * SEQ + s)) * DMODEL + d] = (__bf16)(os[j][r] / lrow[r]);
    }
  }
}

// ---------------------------------------------------------------------------
// Host launcher
// ---------------------------------------------------------------------------
extern "C" void kernel_launch(void* const* d_in, const int* in_sizes, int n_in,
                              void* d_out, int out_size, void* d_ws, size_t ws_size,
                              hipStream_t stream) {
  (void)in_sizes; (void)n_in; (void)out_size; (void)ws_size;
  const float* q  = (const float*)d_in[0];
  const float* k  = (const float*)d_in[1];
  const float* v  = (const float*)d_in[2];
  const float* Wq = (const float*)d_in[3];
  const float* Wk = (const float*)d_in[4];
  const float* Wv = (const float*)d_in[5];
  const float* Wo = (const float*)d_in[6];

  const size_t QKV = (size_t)MROWS * DMODEL;   // 4,194,304 elems
  const size_t WSZ = (size_t)DMODEL * DMODEL;  // 1,048,576 elems

  __bf16* ws = (__bf16*)d_ws;
  __bf16* qb   = ws;
  __bf16* kb   = qb + QKV;
  __bf16* vb   = kb + QKV;
  __bf16* wqb  = vb + QKV;
  __bf16* wkb  = wqb + WSZ;
  __bf16* wvb  = wkb + WSZ;
  __bf16* wob  = wvb + WSZ;
  __bf16* Qh   = wob + WSZ;                    // [B,H,S,dk]
  __bf16* Kh   = Qh + QKV;                     // [B,H,S,dk]
  __bf16* Vt   = Kh + QKV;                     // [B,H,dk,S]
  __bf16* attn = Vt + QKV;                     // [B,S,D]

  const int cvtThreads = 256;
  auto cvt = [&](const float* src, __bf16* dst, size_t n) {
    cvt_f32_to_bf16<<<(int)((n + cvtThreads - 1) / cvtThreads), cvtThreads, 0, stream>>>(src, dst, (int)n);
  };
  cvt(q, qb, QKV); cvt(k, kb, QKV); cvt(v, vb, QKV);
  cvt(Wq, wqb, WSZ); cvt(Wk, wkb, WSZ); cvt(Wv, wvb, WSZ); cvt(Wo, wob, WSZ);

  // GEMM grid: (M/32)*(N/64) waves / 8 waves per block
  const int gemmBlocks = ((MROWS / 32) * (DMODEL / 64)) / 8;   // 256
  gemm_bf16_wmma<<<gemmBlocks, 256, 0, stream>>>(qb, wqb, Qh, MROWS, DMODEL, DMODEL, 0);
  gemm_bf16_wmma<<<gemmBlocks, 256, 0, stream>>>(kb, wkb, Kh, MROWS, DMODEL, DMODEL, 0);
  gemm_bf16_wmma<<<gemmBlocks, 256, 0, stream>>>(vb, wvb, Vt, MROWS, DMODEL, DMODEL, 1);

  // Attention: one (b,h) per block, 8 q-tiles per block -> 32*16 = 512 blocks
  const int attnBlocks = (BATCH * HEADS) * (SEQ / 128);        // 512
  attn_wmma<<<attnBlocks, 256, 0, stream>>>(Qh, Kh, Vt, attn);

  // Final projection -> fp32 d_out
  gemm_bf16_wmma<<<gemmBlocks, 256, 0, stream>>>(attn, wob, d_out, MROWS, DMODEL, DMODEL, 2);
}
